// MS_QCR_Model_89215060672688
// MI455X (gfx1250) — matrix operand order repaired and checked by reading, not compiled, not run on hardware
//
#include <hip/hip_runtime.h>

typedef __attribute__((ext_vector_type(16))) _Float16 v16h;
typedef __attribute__((ext_vector_type(8)))  float    v8f;

// ---------------------------------------------------------------------------
// Setup kernel (1 wave32): build the effective quadratic-form matrix
//   A = Re( U^H * diag(sum_w r_w Z_w) * U ) restricted to the bit2==0 subspace
// by simulating the 3-layer reservoir circuit on all 32 basis columns.
// Output is written directly in the per-lane WMMA B-operand layout:
//   Aop[lane*16 + j] = (lane < 16) ? A[K=j][n=lane] : 0   (K=16..31 zero pad)
// so the main kernel fetches its whole B operand with one 32-byte load.
// ---------------------------------------------------------------------------
__global__ __launch_bounds__(32) void qcr_setup(
    const float* __restrict__ rw,    // reservoir_weights (3,5,5)
    const float* __restrict__ rdw,   // readout_w (1,5)
    _Float16* __restrict__ Aop)      // out: 512 halfs (32 lanes x 16)
{
  __shared__ float Re[32 * 32];     // Re[col*32 + s] = Re U[s][col]
  __shared__ float Im[32 * 32];
  __shared__ float dl[32];
  const int col = threadIdx.x;      // each lane owns one column of U

  #pragma unroll 1
  for (int s = 0; s < 32; ++s) {
    Re[col * 32 + s] = (s == col) ? 1.0f : 0.0f;
    Im[col * 32 + s] = 0.0f;
  }

  for (int layer = 0; layer < 3; ++layer) {
    for (int q = 0; q < 5; ++q) {
      const int bq = 1 << (4 - q);  // wire0 is MSB of the flat state index
      // RX(rw[layer][q][0]):  [[c, -i s], [-i s, c]]
      {
        float t = rw[layer * 25 + q * 5 + 0];
        float c = cosf(0.5f * t), s = sinf(0.5f * t);
        #pragma unroll 1
        for (int p = 0; p < 32; ++p) {
          if (p & bq) continue;
          int p1 = p | bq;
          float a0r = Re[col * 32 + p],  a0i = Im[col * 32 + p];
          float a1r = Re[col * 32 + p1], a1i = Im[col * 32 + p1];
          Re[col * 32 + p]  = c * a0r + s * a1i;
          Im[col * 32 + p]  = c * a0i - s * a1r;
          Re[col * 32 + p1] = c * a1r + s * a0i;
          Im[col * 32 + p1] = c * a1i - s * a0r;
        }
      }
      // RY(rw[layer][q][1]):  [[c, -s], [s, c]] (real)
      {
        float t = rw[layer * 25 + q * 5 + 1];
        float c = cosf(0.5f * t), s = sinf(0.5f * t);
        #pragma unroll 1
        for (int p = 0; p < 32; ++p) {
          if (p & bq) continue;
          int p1 = p | bq;
          float a0r = Re[col * 32 + p],  a0i = Im[col * 32 + p];
          float a1r = Re[col * 32 + p1], a1i = Im[col * 32 + p1];
          Re[col * 32 + p]  = c * a0r - s * a1r;
          Im[col * 32 + p]  = c * a0i - s * a1i;
          Re[col * 32 + p1] = s * a0r + c * a1r;
          Im[col * 32 + p1] = s * a0i + c * a1i;
        }
      }
    }
    // ZZ entanglers: amp *= exp(-0.5i * th * z_i z_j)
    for (int i = 0; i < 5; ++i) {
      for (int j = i + 1; j < 5; ++j) {
        float th = rw[layer * 25 + i * 5 + j];
        float c = cosf(0.5f * th), s = sinf(0.5f * th);
        int bi = 1 << (4 - i), bj = 1 << (4 - j);
        #pragma unroll 1
        for (int p = 0; p < 32; ++p) {
          float zz = ((p & bi) ? -1.f : 1.f) * ((p & bj) ? -1.f : 1.f);
          float pr = c, pi = -zz * s;
          float ar = Re[col * 32 + p], ai = Im[col * 32 + p];
          Re[col * 32 + p] = ar * pr - ai * pi;
          Im[col * 32 + p] = ar * pi + ai * pr;
        }
      }
    }
  }

  // Diagonal observable d(s) = sum_w readout_w[w] * z_w(s)
  {
    float acc = 0.f;
    #pragma unroll
    for (int w = 0; w < 5; ++w)
      acc += rdw[w] * (((col >> (4 - w)) & 1) ? -1.f : 1.f);
    dl[col] = acc;
  }
  __syncthreads();

  // Zero the K=16..31 pad rows of the B operand (lanes 16..31 of main kernel)
  #pragma unroll
  for (int t = 0; t < 8; ++t) Aop[256 + col * 8 + t] = (_Float16)0.f;

  // A(ck,cn) = sum_s d(s) * (ReU[s][fk]*ReU[s][fn] + ImU[s][fk]*ImU[s][fn])
  // compact index c (4 bits i,j,l,m) -> full 5-bit index with bit2(=qubit2)=0
  #pragma unroll 1
  for (int t = 0; t < 8; ++t) {
    int E = col * 8 + t;
    int ck = E >> 4, cn = E & 15;
    int fk = ((ck >> 2) << 3) | (ck & 3);
    int fn = ((cn >> 2) << 3) | (cn & 3);
    float acc = 0.f;
    #pragma unroll 1
    for (int s = 0; s < 32; ++s)
      acc += dl[s] * (Re[fk * 32 + s] * Re[fn * 32 + s] +
                      Im[fk * 32 + s] * Im[fn * 32 + s]);
    Aop[cn * 16 + ck] = (_Float16)acc;   // B-operand layout: lane=cn, half=ck
  }
}

// ---------------------------------------------------------------------------
// Main kernel: 8 waves/block, each wave handles 32 distinct batch elements
// (one per lane). Front-end -> 16-dim real product state row per element,
// staged in LDS; then two v_wmma_f32_16x16x32_f16 tiles (shared B operand)
// compute Y = S*A; output = rowdot(S,Y) + readout_b.
// ---------------------------------------------------------------------------
#define SPITCH 20   // LDS row pitch in floats: 80B = 16B-aligned, bank-clean

__global__ __launch_bounds__(256) void qcr_main(
    const float* __restrict__ xs,  const float* __restrict__ xl,
    const float* __restrict__ csw, const float* __restrict__ csb,
    const float* __restrict__ lsw, const float* __restrict__ lsb,
    const float* __restrict__ clw, const float* __restrict__ clb,
    const float* __restrict__ llw, const float* __restrict__ llb,
    const float* __restrict__ rob,
    const _Float16* __restrict__ Aop,
    float* __restrict__ out, int Btot)
{
  __shared__ float Slds[8][32][SPITCH];      // S rows, f32
  __shared__ float Plds[8][2][16][SPITCH];   // partial products per subtile
  const int wv   = threadIdx.x >> 5;
  const int lane = threadIdx.x & 31;
  const int g    = lane >> 4;                // half-wave
  const int n    = lane & 15;
  const int base = (blockIdx.x * 8 + wv) * 32;
  const int e    = base + lane;              // one element per lane
  const int ee   = (e < Btot) ? e : (Btot - 1);

  // ---- short branch: conv1d(k=3,pad=1) -> ReLU -> linear(20->2) ----
  float xp[7];
  xp[0] = 0.f; xp[6] = 0.f;
  #pragma unroll
  for (int k = 0; k < 5; ++k) xp[1 + k] = xs[ee * 5 + k];
  float fs0 = lsb[0], fs1 = lsb[1];
  #pragma unroll
  for (int c = 0; c < 4; ++c) {
    float b  = csb[c];
    float w0 = csw[c * 3 + 0], w1 = csw[c * 3 + 1], w2 = csw[c * 3 + 2];
    #pragma unroll
    for (int p = 0; p < 5; ++p) {
      float h = fmaxf(b + w0 * xp[p] + w1 * xp[p + 1] + w2 * xp[p + 2], 0.f);
      fs0 += lsw[0 * 20 + c * 5 + p] * h;
      fs1 += lsw[1 * 20 + c * 5 + p] * h;
    }
  }

  // ---- long branch: conv1d(k=5,pad=2) -> ReLU -> maxpool2 -> linear(40->2) --
  float yp[24];
  yp[0] = yp[1] = 0.f; yp[22] = yp[23] = 0.f;
  #pragma unroll
  for (int k = 0; k < 20; ++k) yp[2 + k] = xl[ee * 20 + k];
  float fl0 = llb[0], fl1 = llb[1];
  #pragma unroll
  for (int c = 0; c < 4; ++c) {
    float b  = clb[c];
    float w0 = clw[c * 5 + 0], w1 = clw[c * 5 + 1], w2 = clw[c * 5 + 2];
    float w3 = clw[c * 5 + 3], w4 = clw[c * 5 + 4];
    #pragma unroll
    for (int u = 0; u < 10; ++u) {
      int p0 = 2 * u;
      float h0 = b + w0*yp[p0]   + w1*yp[p0+1] + w2*yp[p0+2] + w3*yp[p0+3] + w4*yp[p0+4];
      float h1 = b + w0*yp[p0+1] + w1*yp[p0+2] + w2*yp[p0+3] + w3*yp[p0+4] + w4*yp[p0+5];
      float pm = fmaxf(fmaxf(h0, 0.f), fmaxf(h1, 0.f));
      fl0 += llw[0 * 40 + c * 10 + u] * pm;
      fl1 += llw[1 * 40 + c * 10 + u] * pm;
    }
  }

  // ---- angles -> 16-dim real product state row, staged to LDS ----
  float q0c = __cosf(0.5f * fs0), q0s = __sinf(0.5f * fs0);
  float q1c = __cosf(0.5f * fs1), q1s = __sinf(0.5f * fs1);
  float q3c = __cosf(0.5f * fl0), q3s = __sinf(0.5f * fl0);
  float q4c = __cosf(0.5f * fl1), q4s = __sinf(0.5f * fl1);
  float p01[4] = {q0c * q1c, q0c * q1s, q0s * q1c, q0s * q1s};
  float p34[4] = {q3c * q4c, q3c * q4s, q3s * q4c, q3s * q4s};
  #pragma unroll
  for (int hi = 0; hi < 4; ++hi)
    #pragma unroll
    for (int lo = 0; lo < 4; ++lo)
      Slds[wv][lane][hi * 4 + lo] = p01[hi] * p34[lo];
  __syncthreads();

  // ---- B operand: one 32-byte load in exact per-lane layout ----
  v16h bm = ((const v16h*)Aop)[lane];

  // ---- two 16x16 WMMA tiles (elements base+t*16 .. base+t*16+15) ----
  #pragma unroll
  for (int t = 0; t < 2; ++t) {
    const float* srow = &Slds[wv][t * 16 + n][0];  // 16B-aligned
    v16h av;
    #pragma unroll
    for (int j = 0; j < 8; ++j) {
      av[j]     = (_Float16)(g ? 0.f : srow[j]);
      av[j + 8] = (_Float16)(g ? 0.f : srow[j + 8]);
    }
    v8f cz = {0.f, 0.f, 0.f, 0.f, 0.f, 0.f, 0.f, 0.f};
    v8f d = __builtin_amdgcn_wmma_f32_16x16x32_f16(
        /*neg_a=*/false, av, /*neg_b=*/false, bm,
        /*c_mod=*/(short)0, cz, /*reuse_a=*/false, /*reuse_b=*/false);
    // D layout: VGPR r, lane group g, col n -> Y[g*8+r][n]
    #pragma unroll
    for (int r = 0; r < 8; ++r)
      Plds[wv][t][g * 8 + r][n] = Slds[wv][t * 16 + g * 8 + r][n] * d[r];
  }
  __syncthreads();

  // ---- final reduction: lane (t=g, row=n) sums its element's 16 partials ---
  float o = rob[0];
  #pragma unroll
  for (int nn = 0; nn < 16; ++nn) o += Plds[wv][g][n][nn];
  if (e < Btot) out[e] = o;
}

extern "C" void kernel_launch(void* const* d_in, const int* in_sizes, int n_in,
                              void* d_out, int out_size, void* d_ws, size_t ws_size,
                              hipStream_t stream) {
  const float* xs  = (const float*)d_in[0];   // x_short (B,1,5)
  const float* xl  = (const float*)d_in[1];   // x_long  (B,1,20)
  const float* csw = (const float*)d_in[2];   // conv_s_w (4,1,3)
  const float* csb = (const float*)d_in[3];   // conv_s_b (4)
  const float* lsw = (const float*)d_in[4];   // lin_s_w (2,20)
  const float* lsb = (const float*)d_in[5];   // lin_s_b (2)
  const float* clw = (const float*)d_in[6];   // conv_l_w (4,1,5)
  const float* clb = (const float*)d_in[7];   // conv_l_b (4)
  const float* llw = (const float*)d_in[8];   // lin_l_w (2,40)
  const float* llb = (const float*)d_in[9];   // lin_l_b (2)
  const float* rw  = (const float*)d_in[10];  // reservoir_weights (3,5,5)
  const float* row = (const float*)d_in[11];  // readout_w (1,5)
  const float* rob = (const float*)d_in[12];  // readout_b (1)
  float* outp = (float*)d_out;

  _Float16* Aop = (_Float16*)d_ws;            // 512 halfs = 1 KB scratch
  const int Btot = in_sizes[0] / 5;

  qcr_setup<<<1, 32, 0, stream>>>(rw, row, Aop);

  const int grid = (Btot + 255) / 256;        // 32 elems/wave * 8 waves/block
  qcr_main<<<grid, 256, 0, stream>>>(xs, xl, csw, csb, lsw, lsb,
                                     clw, clb, llw, llb, rob, Aop,
                                     outp, Btot);
}
